// EdgeDetection_33328946217042
// MI455X (gfx1250) — matrix operand order repaired and checked by reading, not compiled, not run on hardware
//
#include <hip/hip_runtime.h>
#include <hip/hip_bf16.h>

// ---------------------------------------------------------------------------
// Types for CDNA5 WMMA (wave32): A/B = 16x32 / 32x16 f16 (v16h), C/D = 16x16 f32 (v8f)
// ---------------------------------------------------------------------------
typedef __attribute__((ext_vector_type(16))) _Float16 v16h;
typedef __attribute__((ext_vector_type(8)))  float    v8f;

#define ACT_NONE  0
#define ACT_RELU6 1
#define ACT_SIG   2

#define MAXK1 960   // largest Cin of any 1x1 (irs[14..16] proj): 16 oc x 960 k x 2B = 30 KB LDS
#define MAXK3 448   // largest Cin*9 of any 3x3 (erf w3: 48*9=432), padded to 32

// All tensors in this net are < 2^31 elements (max ~92M), so 32-bit unsigned
// offsets are safe everywhere: avoids v_mul_u64/64-bit add chains.

// ---------------------------------------------------------------------------
// 1x1 conv as GEMM: D[oc, pix] = W[oc, cin] * X[cin, pix] (+ epilogue).
// Whole 16 x Kp weight slab staged in LDS once (f32->f16); K-loop has no
// barriers and no weight traffic.  Each wave: 16 oc x 32 pixels, two N-tiles
// share one A fragment -> 2 WMMAs/step.  Full chunks: unguarded loads with
// wave-uniform k*M offsets; the (<=1) tail chunk clamps k and zeroes via
// select.  Epilogue: clamped-index batched loads under uniform null checks;
// only the stores are predicated.  EXEC all-ones at every WMMA.
// ---------------------------------------------------------------------------
__global__ __launch_bounds__(256) void conv1x1_wmma(
    const float* __restrict__ in, const float* __restrict__ wgt,
    const float* __restrict__ bias, const float* __restrict__ bns,
    const float* __restrict__ bnb, const float* __restrict__ res,
    float* __restrict__ out, int Cin, int Cout, int M, int act)
{
  __shared__ _Float16 wtile[16 * MAXK1];

  const int lane = threadIdx.x & 31;
  const int wave = threadIdx.x >> 5;
  const int b    = blockIdx.z;
  const int ocBase  = blockIdx.y * 16;
  const int pixBase = (blockIdx.x * 8 + wave) * 32;

  const int n    = lane & 15;
  const int half = lane >> 4;
  const int K    = Cin;
  const int Kp   = (K + 31) & ~31;

  // ---- stage whole weight slab [16][Kp] once (m-major: no runtime div) ----
  for (int m = 0; m < 16; ++m) {
    int oc = ocBase + m;
    const float* wr = wgt + (unsigned)oc * (unsigned)K;
    for (int k = threadIdx.x; k < Kp; k += 256)
      wtile[m * Kp + k] = (_Float16)((oc < Cout && k < K) ? wr[k] : 0.0f);
  }
  __syncthreads();

  const int  pix0 = pixBase + n;
  const int  pix1 = pixBase + 16 + n;
  const bool pv0  = pix0 < M;
  const bool pv1  = pix1 < M;
  const unsigned p0c = pv0 ? (unsigned)pix0 : 0u;
  const unsigned p1c = pv1 ? (unsigned)pix1 : 0u;

  const unsigned uM  = (unsigned)M;
  const unsigned inB = (unsigned)b * (unsigned)Cin * uM;
  const unsigned base0 = inB + (unsigned)(half * 16) * uM + p0c;
  const unsigned base1 = inB + (unsigned)(half * 16) * uM + p1c;

  v8f acc0 = {}, acc1 = {};
  const int wrow = n * Kp;   // this lane's A row base in LDS

  for (int kb = 0; kb < Kp; kb += 32) {
    v16h afrag, bf0, bf1;
#pragma unroll
    for (int j = 0; j < 16; ++j) {
      // A (16x32 f16) layout: lanes<16 hold K{0-15}, lanes>=16 K{16-31} (matches B)
      int ka = half * 16 + j;
      afrag[j] = wtile[wrow + kb + ka];
    }
    if (kb + 32 <= K) {           // full chunk: unguarded, wave-uniform k offsets
#pragma unroll
      for (int j = 0; j < 16; ++j) {
        unsigned ko = (unsigned)(kb + j) * uM;   // uniform (half-term folded in base)
        bf0[j] = (_Float16)in[base0 + ko];
        bf1[j] = (_Float16)in[base1 + ko];
      }
    } else {                      // tail chunk (<= once): clamp + select, branchless
#pragma unroll
      for (int j = 0; j < 16; ++j) {
        int  kg = kb + half * 16 + j;
        bool kv = kg < K;
        unsigned ko = (unsigned)(kv ? kg : 0) * uM;
        float x0 = in[inB + ko + p0c];
        float x1 = in[inB + ko + p1c];
        bf0[j] = (_Float16)(kv ? x0 : 0.0f);
        bf1[j] = (_Float16)(kv ? x1 : 0.0f);
      }
    }

    acc0 = __builtin_amdgcn_wmma_f32_16x16x32_f16(false, afrag, false, bf0,
                                                  (short)0, acc0, false, false);
    acc1 = __builtin_amdgcn_wmma_f32_16x16x32_f16(false, afrag, false, bf1,
                                                  (short)0, acc1, false, false);

    if (kb + 32 < Kp)             // warm next activation rows
      __builtin_prefetch(&in[base0 + (unsigned)(kb + 32) * uM], 0, 0);
  }

  // ---- epilogue: batched clamped loads (uniform null checks), guarded stores
  unsigned occ[8], ob[8];
  bool ocv[8];
  float biasv[8], bnsv[8], bnbv[8], r0[8], r1[8];
#pragma unroll
  for (int r = 0; r < 8; ++r) {
    int oc = ocBase + r + half * 8;
    ocv[r] = oc < Cout;
    occ[r] = ocv[r] ? (unsigned)oc : (unsigned)(Cout - 1);
    ob[r]  = ((unsigned)b * (unsigned)Cout + occ[r]) * uM;
  }
  if (bias) {
#pragma unroll
    for (int r = 0; r < 8; ++r) biasv[r] = bias[occ[r]];
  } else {
#pragma unroll
    for (int r = 0; r < 8; ++r) biasv[r] = 0.0f;
  }
  if (bns) {
#pragma unroll
    for (int r = 0; r < 8; ++r) { bnsv[r] = bns[occ[r]]; bnbv[r] = bnb[occ[r]]; }
  } else {
#pragma unroll
    for (int r = 0; r < 8; ++r) { bnsv[r] = 1.0f; bnbv[r] = 0.0f; }
  }
  if (res) {
#pragma unroll
    for (int r = 0; r < 8; ++r) { r0[r] = res[ob[r] + p0c]; r1[r] = res[ob[r] + p1c]; }
  } else {
#pragma unroll
    for (int r = 0; r < 8; ++r) { r0[r] = 0.0f; r1[r] = 0.0f; }
  }
#pragma unroll
  for (int r = 0; r < 8; ++r) {
    float v0 = (acc0[r] + biasv[r]) * bnsv[r] + bnbv[r];
    float v1 = (acc1[r] + biasv[r]) * bnsv[r] + bnbv[r];
    if (act == ACT_RELU6) {
      v0 = fminf(fmaxf(v0, 0.0f), 6.0f);
      v1 = fminf(fmaxf(v1, 0.0f), 6.0f);
    } else if (act == ACT_SIG) {
      v0 = 1.0f / (1.0f + __expf(-v0));
      v1 = 1.0f / (1.0f + __expf(-v1));
    }
    v0 += r0[r]; v1 += r1[r];
    if (ocv[r] && pv0) out[ob[r] + (unsigned)pix0] = v0;
    if (ocv[r] && pv1) out[ob[r] + (unsigned)pix1] = v1;
  }
}

// ---------------------------------------------------------------------------
// 3x3 conv (stride runtime, pad 1) as implicit GEMM, K = Cin*9.  Whole weight
// slab LDS-resident; kg/9, rem/3 are compile-time magic multiplies.  B-side
// gathers clamp cin/iy/ix so loads are always valid; zero via select.
// ---------------------------------------------------------------------------
__global__ __launch_bounds__(256) void conv3x3_wmma(
    const float* __restrict__ in, const float* __restrict__ wgt,
    const float* __restrict__ bias, const float* __restrict__ bns,
    const float* __restrict__ bnb, float* __restrict__ out,
    int Cin, int Cout, int IH, int IW, int OH, int OW,
    int stride, int act)
{
  __shared__ _Float16 wtile[16 * MAXK3];

  const int lane = threadIdx.x & 31;
  const int wave = threadIdx.x >> 5;
  const int M  = OH * OW;
  const int K  = Cin * 9;
  const int Kp = (K + 31) & ~31;
  const int b  = blockIdx.z;
  const int ocBase  = blockIdx.y * 16;
  const int pixBase = (blockIdx.x * 8 + wave) * 16;

  for (int m = 0; m < 16; ++m) {
    int oc = ocBase + m;
    const float* wr = wgt + (unsigned)oc * (unsigned)K;
    for (int k = threadIdx.x; k < Kp; k += 256)
      wtile[m * Kp + k] = (_Float16)((oc < Cout && k < K) ? wr[k] : 0.0f);
  }
  __syncthreads();

  const int n    = lane & 15;
  const int half = lane >> 4;

  const int  pix   = pixBase + n;
  const bool pv    = pix < M;
  const int  pix_c = pv ? pix : 0;
  const int  oy    = pix_c / OW;
  const int  ox    = pix_c % OW;
  const unsigned inB = (unsigned)b * (unsigned)Cin * (unsigned)(IH * IW);

  v8f acc = {};
  const int wrowi = n * Kp;

  for (int kb = 0; kb < Kp; kb += 32) {
    v16h afrag, bfrag;
#pragma unroll
    for (int j = 0; j < 16; ++j) {
      int ka = half * 16 + j;
      afrag[j] = wtile[wrowi + kb + ka];

      int  kg = kb + half * 16 + j;
      bool kv = kg < K;
      int  kgc = kv ? kg : 0;
      int  cin = kgc / 9;               // compile-time divisor
      int  rem = kgc - cin * 9;
      int  ky  = rem / 3;
      int  kx  = rem - ky * 3;
      int  iy  = oy * stride - 1 + ky;
      int  ix  = ox * stride - 1 + kx;
      bool valid = kv && iy >= 0 && iy < IH && ix >= 0 && ix < IW;
      int  iyc = min(max(iy, 0), IH - 1);
      int  ixc = min(max(ix, 0), IW - 1);
      float xv = in[inB + ((unsigned)cin * (unsigned)IH + (unsigned)iyc) * (unsigned)IW
                        + (unsigned)ixc];  // always valid addr
      bfrag[j] = (_Float16)(valid ? xv : 0.0f);
    }

    acc = __builtin_amdgcn_wmma_f32_16x16x32_f16(false, afrag, false, bfrag,
                                                 (short)0, acc, false, false);
  }

  // ---- epilogue (batched clamped loads, guarded stores) ----
  unsigned occ[8], ob[8];
  bool ocv[8];
  float biasv[8], bnsv[8], bnbv[8];
#pragma unroll
  for (int r = 0; r < 8; ++r) {
    int oc = ocBase + r + half * 8;
    ocv[r] = oc < Cout;
    occ[r] = ocv[r] ? (unsigned)oc : (unsigned)(Cout - 1);
    ob[r]  = ((unsigned)b * (unsigned)Cout + occ[r]) * (unsigned)M;
  }
  if (bias) {
#pragma unroll
    for (int r = 0; r < 8; ++r) biasv[r] = bias[occ[r]];
  } else {
#pragma unroll
    for (int r = 0; r < 8; ++r) biasv[r] = 0.0f;
  }
  if (bns) {
#pragma unroll
    for (int r = 0; r < 8; ++r) { bnsv[r] = bns[occ[r]]; bnbv[r] = bnb[occ[r]]; }
  } else {
#pragma unroll
    for (int r = 0; r < 8; ++r) { bnsv[r] = 1.0f; bnbv[r] = 0.0f; }
  }
#pragma unroll
  for (int r = 0; r < 8; ++r) {
    float v = (acc[r] + biasv[r]) * bnsv[r] + bnbv[r];
    if (act == ACT_RELU6)    v = fminf(fmaxf(v, 0.0f), 6.0f);
    else if (act == ACT_SIG) v = 1.0f / (1.0f + __expf(-v));
    if (ocv[r] && pv) out[ob[r] + (unsigned)pix] = v;
  }
}

// ---------------------------------------------------------------------------
// Depthwise 3x3 (+BN or +bias, optional ReLU6) — bandwidth-bound VALU kernel.
// ---------------------------------------------------------------------------
__global__ void dwconv3x3(const float* __restrict__ in, const float* __restrict__ wgt,
                          const float* __restrict__ scale, const float* __restrict__ shift,
                          float* __restrict__ out,
                          int B_, int C, int IH, int IW, int OH, int OW,
                          int stride, int dorelu)
{
  unsigned total = (unsigned)B_ * (unsigned)C * (unsigned)(OH * OW);
  for (unsigned i = blockIdx.x * blockDim.x + threadIdx.x; i < total;
       i += gridDim.x * blockDim.x) {
    unsigned ow = i % (unsigned)OW; unsigned t = i / (unsigned)OW;
    unsigned oh = t % (unsigned)OH; t /= (unsigned)OH;
    unsigned c  = t % (unsigned)C;
    unsigned b  = t / (unsigned)C;
    const float* wp = wgt + c * 9u;
    const float* ip = in + (b * (unsigned)C + c) * (unsigned)(IH * IW);
    float acc = 0.0f;
#pragma unroll
    for (int ky = 0; ky < 3; ++ky) {
      int iy = (int)oh * stride - 1 + ky;
      bool yv = (iy >= 0) && (iy < IH);
      int iyc = min(max(iy, 0), IH - 1);
#pragma unroll
      for (int kx = 0; kx < 3; ++kx) {
        int ix = (int)ow * stride - 1 + kx;
        bool v = yv && (ix >= 0) && (ix < IW);
        int ixc = min(max(ix, 0), IW - 1);
        float xv = ip[(unsigned)iyc * (unsigned)IW + (unsigned)ixc];
        acc += (v ? xv : 0.0f) * wp[ky * 3 + kx];
      }
    }
    float v = scale ? (acc * scale[c] + shift[c]) : (acc + shift[c]);
    if (dorelu) v = fminf(fmaxf(v, 0.0f), 6.0f);
    out[i] = v;
  }
}

// ---------------------------------------------------------------------------
// Bilinear upsample (half-pixel centers, matching jax.image.resize) + sigmoid.
// ---------------------------------------------------------------------------
__global__ void upsample_bilinear_sigmoid(const float* __restrict__ in, float* __restrict__ out,
                                          int B_, int ih, int iw, int oh, int ow)
{
  unsigned total = (unsigned)B_ * (unsigned)(oh * ow);
  float sy = (float)ih / (float)oh, sx = (float)iw / (float)ow;
  for (unsigned i = blockIdx.x * blockDim.x + threadIdx.x; i < total;
       i += gridDim.x * blockDim.x) {
    unsigned x = i % (unsigned)ow; unsigned t = i / (unsigned)ow;
    unsigned y = t % (unsigned)oh;
    unsigned b = t / (unsigned)oh;
    float fy = (y + 0.5f) * sy - 0.5f;
    float fx = (x + 0.5f) * sx - 0.5f;
    fy = fminf(fmaxf(fy, 0.0f), (float)(ih - 1));
    fx = fminf(fmaxf(fx, 0.0f), (float)(iw - 1));
    int y0 = (int)fy, x0 = (int)fx;
    int y1 = y0 + 1 < ih ? y0 + 1 : ih - 1;
    int x1 = x0 + 1 < iw ? x0 + 1 : iw - 1;
    float wy = fy - y0, wx = fx - x0;
    const float* ip = in + b * (unsigned)(ih * iw);
    float v00 = ip[(unsigned)y0 * iw + x0], v01 = ip[(unsigned)y0 * iw + x1];
    float v10 = ip[(unsigned)y1 * iw + x0], v11 = ip[(unsigned)y1 * iw + x1];
    float v = (1.0f - wy) * ((1.0f - wx) * v00 + wx * v01)
            +        wy  * ((1.0f - wx) * v10 + wx * v11);
    out[i] = 1.0f / (1.0f + __expf(-v));
  }
}

// ---------------------------------------------------------------------------
// Channel gather for concatenation.
// ---------------------------------------------------------------------------
__global__ void gather_channels(float* __restrict__ dst, const float* __restrict__ src,
                                int B_, int dstC, int dstOff, int srcC, int srcOff,
                                int nC, int HW)
{
  unsigned total = (unsigned)B_ * (unsigned)nC * (unsigned)HW;
  for (unsigned i = blockIdx.x * blockDim.x + threadIdx.x; i < total;
       i += gridDim.x * blockDim.x) {
    unsigned p = i % (unsigned)HW; unsigned t = i / (unsigned)HW;
    unsigned c = t % (unsigned)nC;
    unsigned b = t / (unsigned)nC;
    dst[(b * (unsigned)dstC + (unsigned)dstOff + c) * (unsigned)HW + p] =
        src[(b * (unsigned)srcC + (unsigned)srcOff + c) * (unsigned)HW + p];
  }
}

// ---------------------------------------------------------------------------
// Host-side orchestration
// ---------------------------------------------------------------------------
namespace {

constexpr int BATCH = 4;

struct Arena {
  unsigned char* base; size_t off;
  float* get(size_t nfloats) {
    float* r = (float*)(base + off);
    off += ((nfloats * sizeof(float) + 255) & ~(size_t)255);
    return r;
  }
};

struct BNp { const float* s; const float* b; };
struct IRp {
  bool has_e;
  const float* ew; BNp ebn;
  const float* dww; BNp dbn;
  const float* pw;  BNp pbn;
};
struct HeadP {
  const float* w1; const float* b1; BNp bn1;
  const float* wd; const float* bd;
  const float* w2; const float* b2; BNp bn2;
};

static const int IRC[17][4] = {
  {32,16,1,1},{16,24,2,6},{24,24,1,6},
  {24,32,2,6},{32,32,1,6},{32,32,1,6},
  {32,64,2,6},{64,64,1,6},{64,64,1,6},{64,64,1,6},
  {64,96,1,6},{96,96,1,6},{96,96,1,6},
  {96,160,2,6},{160,160,1,6},{160,160,1,6},{160,320,1,6}};
static const int FEAT_AT[5]  = {2, 5, 9, 12, 16};
static const int HEAD_MID[5] = {48, 64, 128, 192, 512};

static inline void launch_c1(hipStream_t s, const float* in, const float* w,
                             const float* bias, const float* bns, const float* bnb,
                             const float* res, float* out,
                             int Cin, int Cout, int M, int act)
{
  dim3 g((unsigned)((M + 255) / 256), (unsigned)((Cout + 15) / 16), (unsigned)BATCH);
  conv1x1_wmma<<<g, 256, 0, s>>>(in, w, bias, bns, bnb, res, out, Cin, Cout, M, act);
}

static inline void launch_c3(hipStream_t s, const float* in, const float* w,
                             const float* bias, const float* bns, const float* bnb,
                             float* out, int Cin, int Cout,
                             int IH, int IW, int OH, int OW, int stride, int act)
{
  int M = OH * OW;
  dim3 g((unsigned)((M + 127) / 128), (unsigned)((Cout + 15) / 16), (unsigned)BATCH);
  conv3x3_wmma<<<g, 256, 0, s>>>(in, w, bias, bns, bnb, out,
                                 Cin, Cout, IH, IW, OH, OW, stride, act);
}

static inline void launch_dw(hipStream_t s, const float* in, const float* w,
                             const float* scale, const float* shift, float* out,
                             int C, int IH, int IW, int OH, int OW, int stride, int dorelu)
{
  size_t tot = (size_t)BATCH * C * OH * OW;
  size_t nb = (tot + 255) / 256; if (nb > 65535) nb = 65535;
  dwconv3x3<<<(unsigned)nb, 256, 0, s>>>(in, w, scale, shift, out,
                                         BATCH, C, IH, IW, OH, OW, stride, dorelu);
}

static inline void launch_gather(hipStream_t s, float* dst, const float* src,
                                 int dstC, int dstOff, int srcC, int srcOff, int nC, int HW)
{
  size_t tot = (size_t)BATCH * nC * HW;
  size_t nb = (tot + 255) / 256; if (nb > 65535) nb = 65535;
  gather_channels<<<(unsigned)nb, 256, 0, s>>>(dst, src, BATCH, dstC, dstOff, srcC, srcOff, nC, HW);
}

} // namespace

extern "C" void kernel_launch(void* const* d_in, const int* in_sizes, int n_in,
                              void* d_out, int out_size, void* d_ws, size_t ws_size,
                              hipStream_t stream)
{
  (void)in_sizes; (void)n_in; (void)out_size; (void)ws_size;
  const int H0 = 600, W0 = 800, HW0 = H0 * W0;

  // ---- parameter cursor (mirrors setup_inputs() dict insertion order) ----
  int ci = 0;
  auto pop   = [&]() { return (const float*)d_in[ci++]; };
  auto popbn = [&]() { BNp p; p.s = pop(); p.b = pop(); return p; };
  auto popir = [&](bool has_e) {
    IRp r; r.has_e = has_e;
    r.dww = pop(); r.dbn = popbn();
    r.pw  = pop(); r.pbn = popbn();
    if (has_e) { r.ew = pop(); r.ebn = popbn(); }
    else { r.ew = nullptr; r.ebn = BNp{nullptr, nullptr}; }
    return r;
  };

  const float* X = pop();
  const float* stem_w = pop(); BNp stem_bn = popbn();
  IRp irs[17];
  for (int j = 0; j < 17; ++j) irs[j] = popir(IRC[j][3] != 1);
  HeadP heads[5];
  for (int i = 0; i < 5; ++i) {
    heads[i].w1 = pop(); heads[i].b1 = pop(); heads[i].bn1 = popbn();
    heads[i].wd = pop(); heads[i].bd = pop();
    heads[i].w2 = pop(); heads[i].b2 = pop(); heads[i].bn2 = popbn();
  }
  const float *f_w1 = pop(), *f_b1 = pop(), *f_wd = pop(), *f_bd = pop(),
              *f_w2 = pop(), *f_b2 = pop();
  const float* e_w0 = pop(); IRp e_ir = popir(true);
  const float *e_w1 = pop(), *e_b1 = pop(), *e_w2 = pop(), *e_b2 = pop(),
              *e_w3 = pop(), *e_b3 = pop();
  BNp e_bn = popbn();
  IRp p_ir = popir(true);
  const float *p_w1 = pop(), *p_b1 = pop(), *p_w2 = pop(), *p_b2 = pop();

  float* out = (float*)d_out;
  Arena ar{(unsigned char*)d_ws, 0};

  // ---- stem: 3x3 s2 conv + BN + ReLU6 ----
  int curC = 32, curH = 300, curW = 400;
  float* h = ar.get((size_t)BATCH * 32 * curH * curW);
  launch_c3(stream, X, stem_w, nullptr, stem_bn.s, stem_bn.b, h,
            3, 32, H0, W0, curH, curW, 2, ACT_RELU6);

  // ---- 17 inverted-residual blocks; record 5 backbone features ----
  const float* feats[5]; int featC[5], featH[5], featW[5]; int fi = 0;
  for (int j = 0; j < 17; ++j) {
    int inp = IRC[j][0], oup = IRC[j][1], st = IRC[j][2], t = IRC[j][3];
    const float* xin = h;
    const float* cur = h; int cC = inp;
    if (t != 1) {
      int hid = inp * t;
      float* e = ar.get((size_t)BATCH * hid * curH * curW);
      launch_c1(stream, cur, irs[j].ew, nullptr, irs[j].ebn.s, irs[j].ebn.b, nullptr, e,
                inp, hid, curH * curW, ACT_RELU6);
      cur = e; cC = hid;
    }
    int oH = (curH - 1) / st + 1, oW = (curW - 1) / st + 1;
    float* d = ar.get((size_t)BATCH * cC * oH * oW);
    launch_dw(stream, cur, irs[j].dww, irs[j].dbn.s, irs[j].dbn.b, d,
              cC, curH, curW, oH, oW, st, 1);
    float* p = ar.get((size_t)BATCH * oup * oH * oW);
    const float* res = (st == 1 && inp == oup) ? xin : nullptr;
    launch_c1(stream, d, irs[j].pw, nullptr, irs[j].pbn.s, irs[j].pbn.b, res, p,
              cC, oup, oH * oW, ACT_NONE);
    h = p; curC = oup; curH = oH; curW = oW;
    if (fi < 5 && j == FEAT_AT[fi]) { feats[fi] = h; featC[fi] = curC; featH[fi] = curH; featW[fi] = curW; ++fi; }
  }

  // ---- 5 side edge heads -> bilinear upsample + sigmoid into d_out ----
  for (int i = 0; i < 5; ++i) {
    int c = featC[i], m = HEAD_MID[i], Hh = featH[i], Ww = featW[i], M = Hh * Ww;
    float* t1 = ar.get((size_t)BATCH * m * M);
    launch_c1(stream, feats[i], heads[i].w1, heads[i].b1, heads[i].bn1.s, heads[i].bn1.b,
              nullptr, t1, c, m, M, ACT_RELU6);
    float* t2 = ar.get((size_t)BATCH * m * M);
    launch_dw(stream, t1, heads[i].wd, nullptr, heads[i].bd, t2, m, Hh, Ww, Hh, Ww, 1, 1);
    float* t3 = ar.get((size_t)BATCH * M);
    launch_c1(stream, t2, heads[i].w2, heads[i].b2, heads[i].bn2.s, heads[i].bn2.b,
              nullptr, t3, m, 1, M, ACT_NONE);
    size_t tot = (size_t)BATCH * HW0;
    size_t nb = (tot + 255) / 256; if (nb > 65535) nb = 65535;
    upsample_bilinear_sigmoid<<<(unsigned)nb, 256, 0, stream>>>(
        t3, out + (size_t)i * BATCH * HW0, BATCH, Hh, Ww, H0, W0);
  }

  // ---- fuse: concat(x, 5 edges) -> 1x1(8->16)+ReLU6 -> dw3x3+ReLU6 -> 1x1(16->1)+sigmoid
  float* fin = ar.get((size_t)BATCH * 8 * HW0);
  launch_gather(stream, fin, X, 8, 0, 3, 0, 3, HW0);
  for (int i = 0; i < 5; ++i)
    launch_gather(stream, fin, out + (size_t)i * BATCH * HW0, 8, 3 + i, 1, 0, 1, HW0);
  float* ft1 = ar.get((size_t)BATCH * 16 * HW0);
  launch_c1(stream, fin, f_w1, f_b1, nullptr, nullptr, nullptr, ft1, 8, 16, HW0, ACT_RELU6);
  float* ft2 = ar.get((size_t)BATCH * 16 * HW0);
  launch_dw(stream, ft1, f_wd, nullptr, f_bd, ft2, 16, H0, W0, H0, W0, 1, 1);
  launch_c1(stream, ft2, f_w2, f_b2, nullptr, nullptr, nullptr,
            out + (size_t)5 * BATCH * HW0, 16, 1, HW0, ACT_SIG);

  // ---- feature_map = concat(5 edges, fuse) : [B,6,600,800] ----
  float* fm = ar.get((size_t)BATCH * 6 * HW0);
  for (int i = 0; i < 5; ++i)
    launch_gather(stream, fm, out + (size_t)i * BATCH * HW0, 6, i, 1, 0, 1, HW0);
  launch_gather(stream, fm, out + (size_t)5 * BATCH * HW0, 6, 5, 1, 0, 1, HW0);

  // ---- erf: 1x1(6->24); IR(24,24,t=2); 3x stride-2 3x3 (last fused BN+ReLU6)
  float* h0 = ar.get((size_t)BATCH * 24 * HW0);
  launch_c1(stream, fm, e_w0, nullptr, nullptr, nullptr, nullptr, h0, 6, 24, HW0, ACT_NONE);
  float* ee = ar.get((size_t)BATCH * 48 * HW0);
  launch_c1(stream, h0, e_ir.ew, nullptr, e_ir.ebn.s, e_ir.ebn.b, nullptr, ee,
            24, 48, HW0, ACT_RELU6);
  float* ed = ar.get((size_t)BATCH * 48 * HW0);
  launch_dw(stream, ee, e_ir.dww, e_ir.dbn.s, e_ir.dbn.b, ed, 48, H0, W0, H0, W0, 1, 1);
  float* ep = ar.get((size_t)BATCH * 24 * HW0);
  launch_c1(stream, ed, e_ir.pw, nullptr, e_ir.pbn.s, e_ir.pbn.b, h0, ep,
            48, 24, HW0, ACT_NONE);
  float* c1 = ar.get((size_t)BATCH * 32 * 300 * 400);
  launch_c3(stream, ep, e_w1, e_b1, nullptr, nullptr, c1, 24, 32, 600, 800, 300, 400, 2, ACT_NONE);
  float* c2 = ar.get((size_t)BATCH * 48 * 150 * 200);
  launch_c3(stream, c1, e_w2, e_b2, nullptr, nullptr, c2, 32, 48, 300, 400, 150, 200, 2, ACT_NONE);
  float* c3 = ar.get((size_t)BATCH * 64 * 75 * 100);
  launch_c3(stream, c2, e_w3, e_b3, e_bn.s, e_bn.b, c3, 48, 64, 150, 200, 75, 100, 2, ACT_RELU6);

  // ---- erp: IR(64,64,t=2); 1x1(64->32)+b; 1x1(32->3)+b+sigmoid -> pred ----
  const int H3 = 75, W3 = 100, HW3 = H3 * W3;
  float* pe = ar.get((size_t)BATCH * 128 * HW3);
  launch_c1(stream, c3, p_ir.ew, nullptr, p_ir.ebn.s, p_ir.ebn.b, nullptr, pe,
            64, 128, HW3, ACT_RELU6);
  float* pd = ar.get((size_t)BATCH * 128 * HW3);
  launch_dw(stream, pe, p_ir.dww, p_ir.dbn.s, p_ir.dbn.b, pd, 128, H3, W3, H3, W3, 1, 1);
  float* pp = ar.get((size_t)BATCH * 64 * HW3);
  launch_c1(stream, pd, p_ir.pw, nullptr, p_ir.pbn.s, p_ir.pbn.b, c3, pp,
            128, 64, HW3, ACT_NONE);
  float* q1 = ar.get((size_t)BATCH * 32 * HW3);
  launch_c1(stream, pp, p_w1, p_b1, nullptr, nullptr, nullptr, q1, 64, 32, HW3, ACT_NONE);
  launch_c1(stream, q1, p_w2, p_b2, nullptr, nullptr, nullptr,
            out + (size_t)6 * BATCH * HW0, 32, 3, HW3, ACT_SIG);
}
